// Recommender_40896678593052
// MI455X (gfx1250) — matrix operand order repaired and checked by reading, not compiled, not run on hardware
//
#include <hip/hip_runtime.h>
#include <hip/hip_bf16.h>
#include <math.h>

typedef __attribute__((ext_vector_type(16))) _Float16 v16h;
typedef __attribute__((ext_vector_type(8)))  _Float16 v8h;
typedef __attribute__((ext_vector_type(8)))  float    v8f;

#define D_DIM 512
#define NBLK 32          // train rows per block (M side, 2 subtiles of 16)
#define QBLK 64          // question cols per block (N side, 4 subtiles of 16)
#define KB   32          // K slice per WMMA step
#define SA_P 40          // LDS stride (halves) for train tile rows (80B)
#define SB_P 72          // LDS stride (halves) for qnT tile rows (144B)

#define SEG 4096         // bitonic LDS segment (16KB keys + 16KB idx)
#define SORT_THREADS 512

// ---------------------------------------------------------------------------
// gfx1250 async global->LDS copies (ASYNCcnt-tracked; ISA 15.18.3).
// ---------------------------------------------------------------------------
__device__ __forceinline__ unsigned lds_addr(const void* p) {
    return (unsigned)(unsigned long long)p;   // low 32 bits = LDS byte offset
}
__device__ __forceinline__ void async_copy_b128(unsigned lds_off, const void* g) {
    asm volatile("global_load_async_to_lds_b128 %0, %1, off"
                 :: "v"(lds_off), "v"(g) : "memory");
}
__device__ __forceinline__ void async_copy_b64(unsigned lds_off, const void* g) {
    asm volatile("global_load_async_to_lds_b64 %0, %1, off"
                 :: "v"(lds_off), "v"(g) : "memory");
}
__device__ __forceinline__ void wait_async_le2() {   // keep next slice in flight
    asm volatile("s_wait_asynccnt 0x2" ::: "memory");
}
__device__ __forceinline__ void wait_async_le0() {
    asm volatile("s_wait_asynccnt 0x0" ::: "memory");
}

// ---------------------------------------------------------------------------
// Kernel 1: L2-normalize rows of [rows, 512] f32 into f16 with an arbitrary
// output stride pair: y[row*row_stride + i*col_stride].
//   train:    row_stride=D, col_stride=1      (row-major copy)
//   question: row_stride=1, col_stride=Q      (writes transposed qnT[d][q])
// ---------------------------------------------------------------------------
__global__ void normalize_rows_f16(const float* __restrict__ x,
                                   _Float16* __restrict__ y, int D,
                                   size_t row_stride, size_t col_stride) {
    const int row = blockIdx.x;
    const float* xr = x + (size_t)row * D;
    _Float16* yr = y + (size_t)row * row_stride;

    float s = 0.0f;
    for (int i = threadIdx.x; i < D; i += blockDim.x) {
        float v = xr[i];
        s += v * v;
    }
    for (int off = 16; off > 0; off >>= 1) s += __shfl_down(s, off, 32);

    __shared__ float red[8];
    if ((threadIdx.x & 31) == 0) red[threadIdx.x >> 5] = s;
    __syncthreads();
    if (threadIdx.x < 32) {
        float t = (threadIdx.x < (blockDim.x >> 5)) ? red[threadIdx.x] : 0.0f;
        for (int off = 4; off > 0; off >>= 1) t += __shfl_down(t, off, 32);
        if (threadIdx.x == 0) red[0] = t;
    }
    __syncthreads();

    const float inv = 1.0f / fmaxf(sqrtf(red[0]), 1e-8f);
    for (int i = threadIdx.x; i < D; i += blockDim.x)
        yr[(size_t)i * col_stride] = (_Float16)(xr[i] * inv);
}

// ---------------------------------------------------------------------------
// Kernel 2: pad [Q, PADN] beyond column N with -inf / sentinel.
// ---------------------------------------------------------------------------
__global__ void pad_tail(float* __restrict__ keys, int* __restrict__ idxb,
                         int N, int PADN) {
    const int c = N + blockIdx.x * blockDim.x + threadIdx.x;
    const int r = blockIdx.y;
    if (c < PADN) {
        keys[(size_t)r * PADN + c] = -INFINITY;
        idxb[(size_t)r * PADN + c] = 0x7FFFFFFF;
    }
}

// ---------------------------------------------------------------------------
// Kernel 3: WMMA GEMM  scores[n,q] = sum_d tn[n,d]*qnT[d,q].
// M side = train rows, N side = question columns (qnT pre-transposed to [d][q]; A
// fragments are contiguous 16B LDS reads, B fragments are u16 gathers at stride SB_P).
// 256 threads = 8 wave32s; block tile 32(n) x 64(q); wave w -> subtile
// (ns=w&1, qs=w>>1). Double-buffered async global->LDS staging:
// issue slice s+1, s_wait_asynccnt 2 retires slice s (async loads complete
// in order), barrier, 2x ds_load_b128 (A) + 16 u16 LDS reads (B) + v_wmma_f32_16x16x32_f16, barrier.
// ---------------------------------------------------------------------------
__global__ void wmma_scores(const _Float16* __restrict__ tn,   // [N, 512]
                            const _Float16* __restrict__ qnT,  // [512, Q]
                            float* __restrict__ keys,          // [Q, PADN]
                            int* __restrict__ idxb,            // [Q, PADN]
                            int Qtot, int PADN) {
    __shared__ _Float16 sA[2][NBLK * SA_P];   // 2 x 2560 B (train tile)
    __shared__ _Float16 sB[2][KB * SB_P];     // 2 x 4608 B (qnT tile)

    const int t = threadIdx.x;                // 0..255
    const int wave = t >> 5, lane = t & 31;
    const int ns = wave & 1, qs = wave >> 1;
    const int nblk = blockIdx.x * NBLK;
    const int qblk = blockIdx.y * QBLK;

    // staging: A = 256 x b64 (n=t>>3, k=(t&7)*4); B = 256 x b128 (k=t>>3, q=(t&7)*8)
    const _Float16* gA = tn + (size_t)(nblk + (t >> 3)) * D_DIM + (t & 7) * 4;
    const _Float16* gB = qnT + (size_t)(t >> 3) * Qtot + qblk + (t & 7) * 8;
    const unsigned ldsA[2] = {
        lds_addr(&sA[0][(t >> 3) * SA_P + (t & 7) * 4]),
        lds_addr(&sA[1][(t >> 3) * SA_P + (t & 7) * 4]) };
    const unsigned ldsB[2] = {
        lds_addr(&sB[0][(t >> 3) * SB_P + (t & 7) * 8]),
        lds_addr(&sB[1][(t >> 3) * SB_P + (t & 7) * 8]) };

    // fragment read pointers (ISA 7.12.2 layouts)
    const unsigned fa = (unsigned)((ns * 16 + (lane & 15)) * SA_P + (lane >> 4) * 8);
    const unsigned fb = (unsigned)((lane >> 4) * 8 * SB_P + qs * 16 + (lane & 15));
    const _Float16* fragA[2] = { &sA[0][fa], &sA[1][fa] };
    const _Float16* fragB[2] = { &sB[0][fb], &sB[1][fb] };

    // prologue: slice 0 -> buffer 0
    async_copy_b64 (ldsA[0], gA);
    async_copy_b128(ldsB[0], gB);

    v8f acc = {};
    const int S = D_DIM / KB;                 // 16 slices
#pragma unroll 2
    for (int s = 0; s < S; ++s) {
        const int p = s & 1;
        if (s + 1 < S) {
            async_copy_b64 (ldsA[p ^ 1], gA + (size_t)(s + 1) * KB);
            async_copy_b128(ldsB[p ^ 1], gB + (size_t)(s + 1) * KB * Qtot);
            wait_async_le2();                 // slice s resident
        } else {
            wait_async_le0();
        }
        __syncthreads();

        v8h alo = *(const v8h*)(fragA[p]);
        v8h ahi = *(const v8h*)(fragA[p] + 16);
        v16h a, b;
#pragma unroll
        for (int j = 0; j < 8; ++j) {
            a[j] = alo[j]; a[8 + j] = ahi[j];
            b[j] = fragB[p][j * SB_P]; b[8 + j] = fragB[p][(16 + j) * SB_P];
        }

        acc = __builtin_amdgcn_wmma_f32_16x16x32_f16(
            false, a, false, b, (short)0, acc, false, false);

        __syncthreads();                      // readers done before re-staging
    }

    // C layout: M = (lane>>4)*8 + r (train col run), N = lane&15 (question row)
    const int qg    = qblk + qs * 16 + (lane & 15);
    const int nbase = nblk + ns * 16 + (lane >> 4) * 8;
    float4 klo = make_float4(acc[0], acc[1], acc[2], acc[3]);
    float4 khi = make_float4(acc[4], acc[5], acc[6], acc[7]);
    int4   ilo = make_int4(nbase, nbase + 1, nbase + 2, nbase + 3);
    int4   ihi = make_int4(nbase + 4, nbase + 5, nbase + 6, nbase + 7);
    *(float4*)&keys[(size_t)qg * PADN + nbase]     = klo;
    *(float4*)&keys[(size_t)qg * PADN + nbase + 4] = khi;
    *(int4*)&idxb[(size_t)qg * PADN + nbase]       = ilo;
    *(int4*)&idxb[(size_t)qg * PADN + nbase + 4]   = ihi;
}

// ---------------------------------------------------------------------------
// Kernel 4a: bitonic local sort — all stages k=2..SEG in LDS per segment.
// ---------------------------------------------------------------------------
__global__ void bitonic_local_sort(float* __restrict__ keys,
                                   int* __restrict__ idxb, int PADN) {
    __shared__ float sk[SEG];
    __shared__ int   si[SEG];
    const size_t base = (size_t)blockIdx.y * PADN + (size_t)blockIdx.x * SEG;
    const unsigned segbase = blockIdx.x * SEG;

    for (unsigned e = threadIdx.x; e < SEG; e += blockDim.x) {
        sk[e] = keys[base + e];
        si[e] = idxb[base + e];
    }
    __syncthreads();

    for (unsigned k = 2; k <= SEG; k <<= 1) {
        for (unsigned j = k >> 1; j >= 1; j >>= 1) {
            for (unsigned t = threadIdx.x; t < SEG / 2; t += blockDim.x) {
                const unsigned i = ((t & ~(j - 1u)) << 1) | (t & (j - 1u));
                const unsigned l = i | j;
                const bool desc = (((segbase + i) & k) == 0u);
                float ki = sk[i], kl = sk[l];
                if (desc ? (ki < kl) : (ki > kl)) {
                    sk[i] = kl; sk[l] = ki;
                    int ii = si[i]; si[i] = si[l]; si[l] = ii;
                }
            }
            __syncthreads();
        }
    }

    for (unsigned e = threadIdx.x; e < SEG; e += blockDim.x) {
        keys[base + e] = sk[e];
        idxb[base + e] = si[e];
    }
}

// ---------------------------------------------------------------------------
// Kernel 4b: bitonic local merge — sub-passes j=SEG/2..1 of stage k, in LDS.
// ---------------------------------------------------------------------------
__global__ void bitonic_local_merge(float* __restrict__ keys,
                                    int* __restrict__ idxb,
                                    unsigned k, int PADN) {
    __shared__ float sk[SEG];
    __shared__ int   si[SEG];
    const size_t base = (size_t)blockIdx.y * PADN + (size_t)blockIdx.x * SEG;
    const bool desc = (((blockIdx.x * SEG) & k) == 0u);

    for (unsigned e = threadIdx.x; e < SEG; e += blockDim.x) {
        sk[e] = keys[base + e];
        si[e] = idxb[base + e];
    }
    __syncthreads();

    for (unsigned j = SEG >> 1; j >= 1; j >>= 1) {
        for (unsigned t = threadIdx.x; t < SEG / 2; t += blockDim.x) {
            const unsigned i = ((t & ~(j - 1u)) << 1) | (t & (j - 1u));
            const unsigned l = i | j;
            float ki = sk[i], kl = sk[l];
            if (desc ? (ki < kl) : (ki > kl)) {
                sk[i] = kl; sk[l] = ki;
                int ii = si[i]; si[i] = si[l]; si[l] = ii;
            }
        }
        __syncthreads();
    }

    for (unsigned e = threadIdx.x; e < SEG; e += blockDim.x) {
        keys[base + e] = sk[e];
        idxb[base + e] = si[e];
    }
}

// ---------------------------------------------------------------------------
// Kernel 4c: one global bitonic pass (j >= SEG, crosses segments).
// ---------------------------------------------------------------------------
__global__ void bitonic_pass(float* __restrict__ keys, int* __restrict__ idxb,
                             unsigned j, unsigned k, int PADN) {
    const unsigned t = blockIdx.x * blockDim.x + threadIdx.x;   // [0, PADN/2)
    const size_t base = (size_t)blockIdx.y * PADN;
    const unsigned i = ((t & ~(j - 1u)) << 1) | (t & (j - 1u));
    const unsigned l = i | j;

    float ki = keys[base + i];
    float kl = keys[base + l];
    const bool desc = ((i & k) == 0u);
    if (desc ? (ki < kl) : (ki > kl)) {
        int ii = idxb[base + i];
        int il = idxb[base + l];
        keys[base + i] = kl;  keys[base + l] = ki;
        idxb[base + i] = il;  idxb[base + l] = ii;
    }
}

// ---------------------------------------------------------------------------
// Kernel 5: gather sorted [Q, N] scores + indices into d_out.
// ---------------------------------------------------------------------------
__global__ void gather_out(const float* __restrict__ keys,
                           const int* __restrict__ idxb,
                           float* __restrict__ out_scores,
                           int* __restrict__ out_idx,
                           int N, int PADN) {
    const int c = blockIdx.x * blockDim.x + threadIdx.x;
    const int r = blockIdx.y;
    if (c < N) {
        out_scores[(size_t)r * N + c] = keys[(size_t)r * PADN + c];
        out_idx[(size_t)r * N + c]    = idxb[(size_t)r * PADN + c];
    }
}

// ---------------------------------------------------------------------------
extern "C" void kernel_launch(void* const* d_in, const int* in_sizes, int n_in,
                              void* d_out, int out_size, void* d_ws, size_t ws_size,
                              hipStream_t stream) {
    const float* question = (const float*)d_in[0];   // [Q, 512] f32
    const float* train    = (const float*)d_in[1];   // [N, 512] f32

    const int D = D_DIM;
    const int Q = in_sizes[0] / D;                   // 1024
    const int N = in_sizes[1] / D;                   // 100000
    int PADN = 1;
    while (PADN < N) PADN <<= 1;                     // 131072

    // Workspace layout
    char* ws = (char*)d_ws;
    size_t off = 0;
    _Float16* qnT = (_Float16*)(ws + off); off += (size_t)Q * D * sizeof(_Float16);
    off = (off + 255) & ~(size_t)255;
    _Float16* tnH = (_Float16*)(ws + off); off += (size_t)N * D * sizeof(_Float16);
    off = (off + 255) & ~(size_t)255;
    float* keys = (float*)(ws + off);      off += (size_t)Q * PADN * sizeof(float);
    int*   idxb = (int*)(ws + off);

    // 1) normalize: train row-major f16; question transposed -> qnT[d][q]
    normalize_rows_f16<<<Q, 256, 0, stream>>>(question, qnT, D, 1, Q);
    normalize_rows_f16<<<N, 256, 0, stream>>>(train, tnH, D, D, 1);

    // 2) pad the sort buffers
    {
        dim3 grid((PADN - N + 255) / 256, Q);
        pad_tail<<<grid, 256, 0, stream>>>(keys, idxb, N, PADN);
    }

    // 3) WMMA GEMM with double-buffered async LDS staging (N%32==0, Q%64==0)
    {
        dim3 grid(N / NBLK, Q / QBLK);
        wmma_scores<<<grid, 256, 0, stream>>>(tnH, qnT, keys, idxb, Q, PADN);
    }

    // 4) bitonic sort, descending per row (LDS segments + sparse global passes)
    {
        dim3 lgrid(PADN / SEG, Q);
        bitonic_local_sort<<<lgrid, SORT_THREADS, 0, stream>>>(keys, idxb, PADN);
        dim3 ggrid((unsigned)(PADN / 2 / 256), (unsigned)Q);
        for (unsigned k = SEG * 2; k <= (unsigned)PADN; k <<= 1) {
            for (unsigned j = k >> 1; j >= SEG; j >>= 1)
                bitonic_pass<<<ggrid, 256, 0, stream>>>(keys, idxb, j, k, PADN);
            bitonic_local_merge<<<lgrid, SORT_THREADS, 0, stream>>>(keys, idxb, k, PADN);
        }
    }

    // 5) gather into d_out: [Q*N] f32 scores then [Q*N] i32 indices
    {
        float* out_scores = (float*)d_out;
        int*   out_idx    = (int*)((float*)d_out + (size_t)Q * N);
        dim3 grid((N + 255) / 256, Q);
        gather_out<<<grid, 256, 0, stream>>>(keys, idxb, out_scores, out_idx, N, PADN);
    }
}
